// Decoder_9509057593545
// MI455X (gfx1250) — compile-verified, matrix operand order checked by reading, not run on hardware
//
#include <hip/hip_runtime.h>
#include <hip/hip_bf16.h>
#include <math.h>

// ---------------- problem constants ----------------
#define B_   16
#define S_   8
#define TK_  400
#define H_   256
#define E_   128
#define V_   50000
#define OOV_ 50
#define H2_  512      // 2*H
#define NROW (B_*S_*TK_)   // 51200 word-attention rows
#define ROWB (S_*TK_)      // 3200 rows per batch
#define VP_  (V_+OOV_)     // 50050

// ---------------- d_out layout (floats) ----------------
#define FD_OFF   0                        // [B, V+OOV] = 800800
#define H_OFF    (FD_OFF + B_*VP_)        // 800800
#define C_OFF    (H_OFF + B_*H_)          // 804896
#define CT_OFF   (C_OFF + B_*H_)          // 808992
#define AT_OFF   (CT_OFF + B_*H2_)        // 817184
#define PG_OFF   (AT_OFF + B_*ROWB)       // 868384
#define COV_OFF  (PG_OFF + B_)            // 868400

// ---------------- workspace layout (float offsets) ----------------
#define X_WS     0                        // [B,E]      2048
#define S_WS     (X_WS + B_*E_)           // s_t_hat [B,H2] 8192
#define DFS_WS   (S_WS + B_*H2_)          // dec_fea section [B,H2]
#define DFW_WS   (DFS_WS + B_*H2_)        // dec_fea word    [B,H2]
#define SCS_WS   (DFW_WS + B_*H2_)        // sc_sec [B,S]
#define SCW_WS   (SCS_WS + B_*S_)         // sc_w   [B,S*TK] 51200
#define OUT1_WS  (SCW_WS + B_*ROWB)       // out1 [B,H] 4096
#define WPACK_WS (OUT1_WS + B_*H_)        // bf16 packed Wh_w (512KB)

typedef __attribute__((ext_vector_type(16))) __bf16 v16bf;
typedef __attribute__((ext_vector_type(8)))  float  v8f;

__device__ __forceinline__ float sigm(float x) { return 1.0f / (1.0f + __expf(-x)); }

// pack 16 floats (two 8-float K runs, 16 apart) into a bf16 A/B fragment
__device__ __forceinline__ v16bf cvt16(float4 p0, float4 p1, float4 q0, float4 q1) {
    v16bf a;
    a[0]=(__bf16)p0.x; a[1]=(__bf16)p0.y; a[2]=(__bf16)p0.z; a[3]=(__bf16)p0.w;
    a[4]=(__bf16)p1.x; a[5]=(__bf16)p1.y; a[6]=(__bf16)p1.z; a[7]=(__bf16)p1.w;
    a[8]=(__bf16)q0.x; a[9]=(__bf16)q0.y; a[10]=(__bf16)q0.z; a[11]=(__bf16)q0.w;
    a[12]=(__bf16)q1.x; a[13]=(__bf16)q1.y; a[14]=(__bf16)q1.z; a[15]=(__bf16)q1.w;
    return a;
}

// ================= kernel 0: repack Wh_w -> bf16 WMMA tile blob =================
// layout: Wpack[ntile][kstep][lane][j], j<8 -> K=kstep*32+koff+j, j>=8 -> +16
__global__ void k0_repack(const float* __restrict__ Wh_w, __bf16* __restrict__ Wpack) {
    int idx = blockIdx.x * 256 + threadIdx.x;          // 262144 total
    int j    = idx & 15;
    int lane = (idx >> 4) & 31;
    int ks   = (idx >> 9) & 15;
    int nt   = idx >> 13;
    int K = ks * 32 + ((j < 8) ? 0 : 16) + ((lane >> 4) << 3) + (j & 7);
    int N = nt * 16 + (lane & 15);
    Wpack[idx] = (__bf16)Wh_w[K * H2_ + N];
}

// ================= kernel 1: embed + x-proj + LSTM + dec_fea =================
__global__ void k1_lstm(const int* __restrict__ y, const float* __restrict__ h0,
                        const float* __restrict__ c0, const float* __restrict__ c_t_1,
                        const float* __restrict__ emb_tab,
                        const float* __restrict__ W_xc, const float* __restrict__ b_xc,
                        const float* __restrict__ W_ih, const float* __restrict__ b_ih,
                        const float* __restrict__ W_hh, const float* __restrict__ b_hh,
                        const float* __restrict__ Wd_sec, const float* __restrict__ bd_sec,
                        const float* __restrict__ Wd_w, const float* __restrict__ bd_w,
                        float* __restrict__ out, float* __restrict__ ws) {
    __shared__ float x_lds[B_ * E_];    // 8 KB
    __shared__ float s_lds[B_ * H2_];   // 32 KB
    int tid = threadIdx.x;              // 256 threads
    // phase A: x = [c_t_1, emb] @ W_xc + b_xc
    for (int it = 0; it < B_ * E_ / 256; ++it) {
        int idx = it * 256 + tid;
        int b = idx >> 7, e = idx & 127;
        int tok = y[b];
        float s = b_xc[e];
        for (int k = 0; k < H2_; ++k) s += c_t_1[b * H2_ + k] * W_xc[k * E_ + e];
        for (int k = 0; k < E_; ++k)  s += emb_tab[tok * E_ + k] * W_xc[(H2_ + k) * E_ + e];
        x_lds[idx] = s;
        ws[X_WS + idx] = s;
    }
    __syncthreads();
    // phase B: LSTM cell, gate order i,f,g,o
    for (int it = 0; it < B_ * H_ / 256; ++it) {
        int idx = it * 256 + tid;
        int b = idx >> 8, n = idx & 255;
        float gv[4];
        for (int q = 0; q < 4; ++q) {
            int jj = q * H_ + n;
            float s = b_ih[jj] + b_hh[jj];
            for (int k = 0; k < E_; ++k) s += x_lds[b * E_ + k] * W_ih[k * 4 * H_ + jj];
            for (int k = 0; k < H_; ++k) s += h0[b * H_ + k] * W_hh[k * 4 * H_ + jj];
            gv[q] = s;
        }
        float c = sigm(gv[1]) * c0[idx] + sigm(gv[0]) * tanhf(gv[2]);
        float h = sigm(gv[3]) * tanhf(c);
        out[H_OFF + idx] = h;
        out[C_OFF + idx] = c;
        s_lds[b * H2_ + n] = h;
        s_lds[b * H2_ + H_ + n] = c;
        ws[S_WS + b * H2_ + n] = h;
        ws[S_WS + b * H2_ + H_ + n] = c;
    }
    __syncthreads();
    // phase C: dec_fea for both attention levels
    for (int it = 0; it < B_ * H2_ / 256; ++it) {
        int idx = it * 256 + tid;
        int b = idx >> 9, n = idx & 511;
        float s1 = bd_sec[n], s2 = bd_w[n];
        for (int k = 0; k < H2_; ++k) {
            float sv = s_lds[b * H2_ + k];
            s1 += sv * Wd_sec[k * H2_ + n];
            s2 += sv * Wd_w[k * H2_ + n];
        }
        ws[DFS_WS + idx] = s1;
        ws[DFW_WS + idx] = s2;
    }
}

// ================= kernel 2: section attention scores =================
__global__ void k2_section(const float* __restrict__ sec_out, const float* __restrict__ Wh_sec,
                           const float* __restrict__ v_sec, float* __restrict__ ws) {
    __shared__ float red[256];
    int bs = blockIdx.x;            // 0..127 = b*8+s
    int b = bs >> 3;
    int tid = threadIdx.x;
    const float* row = sec_out + bs * H2_;
    float part = 0.f;
    for (int col = tid; col < H2_; col += 256) {
        float s = ws[DFS_WS + b * H2_ + col];
        for (int k = 0; k < H2_; ++k) s += row[k] * Wh_sec[k * H2_ + col];
        part += tanhf(s) * v_sec[col];
    }
    red[tid] = part;
    __syncthreads();
    for (int off = 128; off > 0; off >>= 1) {
        if (tid < off) red[tid] += red[tid + off];
        __syncthreads();
    }
    if (tid == 0) ws[SCS_WS + bs] = red[0];
}

// ================= kernel 3: fused word-attention score GEMM (WMMA) =================
// sc_w[row] = sum_n tanh( (enc @ Wh_w)[row,n] + dec_fea_w[b,n] + cov[row]*Wc_w[n] ) * v_w[n]
// 8 waves/block, 16 rows/wave; A tiles live in registers for the whole block;
// B panels double-buffered through LDS so staging overlaps WMMA.
__global__ void __launch_bounds__(256, 1)
k3_wordscore(const float* __restrict__ enc, const __bf16* __restrict__ Wpack,
             const float* __restrict__ Wc_w, const float* __restrict__ v_w,
             const float* __restrict__ coverage, float* __restrict__ ws) {
    __shared__ __bf16 bpan[2][16 * 32 * 16];  // two N-tile B panels, 2 x 16 KB
    int tid = threadIdx.x;                    // 256 = 8 waves
    int lane = tid & 31, wave = tid >> 5;
    int Mbase = blockIdx.x * 128 + wave * 16;
    int bidx = Mbase / ROWB;
    int koff = (lane >> 4) << 3;              // 0 or 8

    // preload A fragments: 16 rows x 512 K as bf16 (held in registers)
    const float4* A4 = (const float4*)(enc + (size_t)(Mbase + (lane & 15)) * H2_);
    v16bf aF[16];
#pragma unroll
    for (int ks = 0; ks < 16; ++ks) {
        int base = ks * 8 + (koff >> 2);
        aF[ks] = cvt16(A4[base], A4[base + 1], A4[base + 4], A4[base + 5]);
    }
    // coverage values for the 8 D-rows this lane owns
    float covv[8], rowsum[8];
    int rbase = Mbase + ((lane >> 4) << 3);
#pragma unroll
    for (int r = 0; r < 8; ++r) { covv[r] = coverage[rbase + r]; rowsum[r] = 0.f; }

    // stage panel 0
    {
        const uint4* src = (const uint4*)(Wpack);
        uint4* dst = (uint4*)bpan[0];
#pragma unroll
        for (int i = 0; i < 4; ++i) dst[i * 256 + tid] = src[i * 256 + tid];
    }
    for (int nt = 0; nt < 32; ++nt) {
        __syncthreads();                      // panel nt is ready; panel nt-1 fully consumed
        if (nt + 1 < 32) {                    // stage next panel into the other buffer
            const uint4* src = (const uint4*)(Wpack + (nt + 1) * 8192);
            uint4* dst = (uint4*)bpan[(nt + 1) & 1];
#pragma unroll
            for (int i = 0; i < 4; ++i) dst[i * 256 + tid] = src[i * 256 + tid];
        }
        const __bf16* cur = bpan[nt & 1];
        v8f acc = {0.f, 0.f, 0.f, 0.f, 0.f, 0.f, 0.f, 0.f};
#pragma unroll
        for (int ks = 0; ks < 16; ++ks) {
            v16bf bF = *(const v16bf*)&cur[(ks * 32 + lane) * 16];
            acc = __builtin_amdgcn_wmma_f32_16x16x32_bf16(
                false, aF[ks], false, bF, (short)0, acc, false, false);
        }
        int col = nt * 16 + (lane & 15);
        float df = ws[DFW_WS + bidx * H2_ + col];
        float wc = Wc_w[col];
        float vw = v_w[col];
#pragma unroll
        for (int r = 0; r < 8; ++r)
            rowsum[r] += tanhf(acc[r] + df + covv[r] * wc) * vw;
    }
    // reduce over the 16 columns held by the half-wave
#pragma unroll
    for (int m = 1; m <= 8; m <<= 1)
#pragma unroll
        for (int r = 0; r < 8; ++r)
            rowsum[r] += __shfl_xor(rowsum[r], m, 32);
    if ((lane & 15) == 0)
#pragma unroll
        for (int r = 0; r < 8; ++r)
            ws[SCW_WS + rbase + r] = rowsum[r];
}

// ================= kernel 4: beta softmax + word softmax + c_t + coverage =================
__global__ void k4_combine(const float* __restrict__ mask, const float* __restrict__ coverage,
                           const float* __restrict__ enc, float* __restrict__ out,
                           float* __restrict__ ws) {
    __shared__ float vals[ROWB];   // 12.8 KB
    __shared__ float red[512];
    __shared__ float betaS[S_];
    int b = blockIdx.x, tid = threadIdx.x;     // 512 threads
    if (tid == 0) {
        float mx = -1e30f;
        for (int s = 0; s < S_; ++s) mx = fmaxf(mx, ws[SCS_WS + b * S_ + s]);
        float sm = 0.f;
        for (int s = 0; s < S_; ++s) { betaS[s] = __expf(ws[SCS_WS + b * S_ + s] - mx); sm += betaS[s]; }
        for (int s = 0; s < S_; ++s) betaS[s] /= sm;
    }
    __syncthreads();
    float lmax = -1e30f;
    for (int t = tid; t < ROWB; t += 512) {
        float v = betaS[t / TK_] * ws[SCW_WS + b * ROWB + t];
        vals[t] = v;
        lmax = fmaxf(lmax, v);
    }
    red[tid] = lmax; __syncthreads();
    for (int off = 256; off > 0; off >>= 1) {
        if (tid < off) red[tid] = fmaxf(red[tid], red[tid + off]);
        __syncthreads();
    }
    float gmax = red[0]; __syncthreads();
    float lsum = 0.f;
    for (int t = tid; t < ROWB; t += 512) {
        float e = __expf(vals[t] - gmax) * mask[b * ROWB + t];
        vals[t] = e; lsum += e;
    }
    red[tid] = lsum; __syncthreads();
    for (int off = 256; off > 0; off >>= 1) {
        if (tid < off) red[tid] += red[tid + off];
        __syncthreads();
    }
    float inv = 1.f / red[0]; __syncthreads();
    for (int t = tid; t < ROWB; t += 512) {
        float a = vals[t] * inv;
        vals[t] = a;
        out[AT_OFF + b * ROWB + t] = a;
        out[COV_OFF + b * ROWB + t] = coverage[b * ROWB + t] + a;
    }
    __syncthreads();
    // c_t: each thread owns one of 512 channels; loads coalesced across threads
    int n = tid;
    const float* ep = enc + (size_t)b * ROWB * H2_ + n;
    float acc = 0.f;
    for (int t = 0; t < ROWB; ++t) acc += vals[t] * ep[(size_t)t * H2_];
    out[CT_OFF + b * H2_ + n] = acc;
}

// ================= kernel 5: out1 GEMM + p_gen =================
__global__ void k5_small(const float* __restrict__ W_out1, const float* __restrict__ b_out1,
                         const float* __restrict__ W_pgen, const float* __restrict__ b_pgen,
                         float* __restrict__ out, float* __restrict__ ws) {
    int tid = threadIdx.x;   // 256
    for (int it = 0; it < B_ * H_ / 256; ++it) {
        int idx = it * 256 + tid;
        int b = idx >> 8, n = idx & 255;
        float s = b_out1[n];
        for (int k = 0; k < H_; ++k)  s += out[H_OFF + b * H_ + k] * W_out1[k * H_ + n];
        for (int k = 0; k < H2_; ++k) s += out[CT_OFF + b * H2_ + k] * W_out1[(H_ + k) * H_ + n];
        ws[OUT1_WS + idx] = s;
    }
    if (tid < B_) {
        int b = tid;
        float s = b_pgen[0];
        for (int k = 0; k < H2_; ++k) s += out[CT_OFF + b * H2_ + k] * W_pgen[k];
        for (int k = 0; k < H2_; ++k) s += ws[S_WS + b * H2_ + k] * W_pgen[H2_ + k];
        for (int k = 0; k < E_; ++k)  s += ws[X_WS + b * E_ + k] * W_pgen[2 * H2_ + k];
        out[PG_OFF + b] = sigm(s);
    }
}

// ================= kernel 6: vocab logits (WMMA, M=16=B) -> fd region =================
__global__ void __launch_bounds__(128, 1)
k6_vocab(const float* __restrict__ W_out2, const float* __restrict__ b_out2,
         float* __restrict__ out, const float* __restrict__ ws) {
    int lane = threadIdx.x & 31, wave = threadIdx.x >> 5;
    int nt = blockIdx.x * 4 + wave;
    if (nt >= V_ / 16) return;
    int koff = (lane >> 4) << 3;
    const float4* A4 = (const float4*)(ws + OUT1_WS + (lane & 15) * H_);
    int col = nt * 16 + (lane & 15);
    v8f acc = {0.f, 0.f, 0.f, 0.f, 0.f, 0.f, 0.f, 0.f};
#pragma unroll
    for (int ks = 0; ks < 8; ++ks) {
        int base = ks * 8 + (koff >> 2);
        v16bf aF = cvt16(A4[base], A4[base + 1], A4[base + 4], A4[base + 5]);
        v16bf bF;
#pragma unroll
        for (int j = 0; j < 16; ++j) {
            int K = ks * 32 + ((j < 8) ? 0 : 16) + koff + (j & 7);
            bF[j] = (__bf16)W_out2[(size_t)K * V_ + col];
        }
        acc = __builtin_amdgcn_wmma_f32_16x16x32_bf16(
            false, aF, false, bF, (short)0, acc, false, false);
    }
    float bias = b_out2[col];
    int rbase = (lane >> 4) << 3;
#pragma unroll
    for (int r = 0; r < 8; ++r)
        out[FD_OFF + (size_t)(rbase + r) * VP_ + col] = acc[r] + bias;
}

// ================= kernel 7: in-place softmax over V, scale by p_gen, zero OOV =================
__global__ void k7_vsoftmax(float* __restrict__ out) {
    __shared__ float red[1024];
    int b = blockIdx.x, tid = threadIdx.x;   // 1024 threads
    float* fd = out + FD_OFF + (size_t)b * VP_;
    float lmax = -1e30f;
    for (int v = tid; v < V_; v += 1024) lmax = fmaxf(lmax, fd[v]);
    red[tid] = lmax; __syncthreads();
    for (int off = 512; off > 0; off >>= 1) {
        if (tid < off) red[tid] = fmaxf(red[tid], red[tid + off]);
        __syncthreads();
    }
    float gmax = red[0]; __syncthreads();
    float lsum = 0.f;
    for (int v = tid; v < V_; v += 1024) lsum += __expf(fd[v] - gmax);
    red[tid] = lsum; __syncthreads();
    for (int off = 512; off > 0; off >>= 1) {
        if (tid < off) red[tid] += red[tid + off];
        __syncthreads();
    }
    float pg = out[PG_OFF + b];
    float scale = pg / red[0];
    for (int v = tid; v < V_; v += 1024) fd[v] = __expf(fd[v] - gmax) * scale;
    for (int v = V_ + tid; v < VP_; v += 1024) fd[v] = 0.f;
}

// ================= kernel 8: pointer scatter-add =================
__global__ void k8_scatter(const int* __restrict__ ebev, float* __restrict__ out) {
    int i = blockIdx.x * 256 + threadIdx.x;  // B*ROWB
    int b = i / ROWB, t = i % ROWB;
    float ad = (1.0f - out[PG_OFF + b]) * out[AT_OFF + b * ROWB + t];
    int idx = ebev[b * ROWB + t];
    atomicAdd(&out[FD_OFF + (size_t)b * VP_ + idx], ad);
}

// ================= host launch =================
extern "C" void kernel_launch(void* const* d_in, const int* in_sizes, int n_in,
                              void* d_out, int out_size, void* d_ws, size_t ws_size,
                              hipStream_t stream) {
    const int*   y        = (const int*)  d_in[0];
    const float* h0       = (const float*)d_in[1];
    const float* c0       = (const float*)d_in[2];
    const float* enc      = (const float*)d_in[3];
    const float* sec      = (const float*)d_in[4];
    const float* mask     = (const float*)d_in[5];
    const float* c_t_1    = (const float*)d_in[6];
    const int*   ebev     = (const int*)  d_in[8];
    const float* coverage = (const float*)d_in[9];
    const float* emb_tab  = (const float*)d_in[11];
    const float* W_xc     = (const float*)d_in[12];
    const float* b_xc     = (const float*)d_in[13];
    const float* W_ih     = (const float*)d_in[14];
    const float* b_ih     = (const float*)d_in[15];
    const float* W_hh     = (const float*)d_in[16];
    const float* b_hh     = (const float*)d_in[17];
    const float* Wh_sec   = (const float*)d_in[18];
    const float* Wd_sec   = (const float*)d_in[19];
    const float* bd_sec   = (const float*)d_in[20];
    const float* v_sec    = (const float*)d_in[21];
    const float* Wh_w     = (const float*)d_in[22];
    const float* Wc_w     = (const float*)d_in[23];
    const float* Wd_w     = (const float*)d_in[24];
    const float* bd_w     = (const float*)d_in[25];
    const float* v_w      = (const float*)d_in[26];
    const float* W_pgen   = (const float*)d_in[27];
    const float* b_pgen   = (const float*)d_in[28];
    const float* W_out1   = (const float*)d_in[29];
    const float* b_out1   = (const float*)d_in[30];
    const float* W_out2   = (const float*)d_in[31];
    const float* b_out2   = (const float*)d_in[32];

    float* out = (float*)d_out;
    float* ws  = (float*)d_ws;
    __bf16* Wpack = (__bf16*)(ws + WPACK_WS);

    k0_repack<<<1024, 256, 0, stream>>>(Wh_w, Wpack);
    k1_lstm<<<1, 256, 0, stream>>>(y, h0, c0, c_t_1, emb_tab, W_xc, b_xc,
                                   W_ih, b_ih, W_hh, b_hh,
                                   Wd_sec, bd_sec, Wd_w, bd_w, out, ws);
    k2_section<<<B_ * S_, 256, 0, stream>>>(sec, Wh_sec, v_sec, ws);
    k3_wordscore<<<NROW / 128, 256, 0, stream>>>(enc, Wpack, Wc_w, v_w, coverage, ws);
    k4_combine<<<B_, 512, 0, stream>>>(mask, coverage, enc, out, ws);
    k5_small<<<1, 256, 0, stream>>>(W_out1, b_out1, W_pgen, b_pgen, out, ws);
    k6_vocab<<<(V_ / 16 + 3) / 4, 128, 0, stream>>>(W_out2, b_out2, out, ws);
    k7_vsoftmax<<<B_, 1024, 0, stream>>>(out);
    k8_scatter<<<B_ * ROWB / 256, 256, 0, stream>>>(ebev, out);
}
